// GromovWassersteinEmbedding_45286135169680
// MI455X (gfx1250) — matrix operand order repaired and checked by reading, not compiled, not run on hardware
//
#include <hip/hip_runtime.h>
#include <stdint.h>

// GW embedding loss for MI455X (gfx1250, wave32, WMMA).
// n = 4096, d = 128. Output: (d_gw, d_w, regularizer) as 3 floats.
//
// Pipeline (all on `stream`):
//  0. zero accumulators (f1,f2,G1,G2,scalars)
//  1. gather embeddings -> bf16 + row energies (fp32) + G = E^T E (ortho)
//  2. trans -> bf16 (+ transposed bf16 copy), row sums, col sums
//  3. cost_self  (E E^T via WMMA bf16) -> cost bf16, fused f-vec + similarity reg
//  4. cost_mutual (E1 E2^T via WMMA)   -> fused d_w = <cost_st, T>  (never stored)
//  5. M2 = T @ cost_t      (WMMA bf16, K=4096) -> fp32 workspace
//  6. <cost_s @ T, M2>     (WMMA bf16, fused dot epilogue) -> gw scalar
//  7. final combine -> d_out[3]
//
// GEMM staging demonstrates BOTH CDNA5 DMA paths with double-buffered LDS:
//   A tile: Tensor Data Mover (TENSOR_LOAD_TO_LDS descriptor DMA, TENSORcnt),
//           hardware pad fields reproduce the bank-conflict-free 80B pitch.
//   B tile: per-lane GLOBAL_LOAD_ASYNC_TO_LDS_B128 (ASYNCcnt).
// Tile t+1 streams in while tile t computes on WMMA.

#define N_PT 4096
#define D_EMB 128
#define EPSV 1e-5f
#define TKS 32
#define KPAD 40   // 80B LDS row pitch: conflict-free 16-lane b128 fragment reads

typedef __attribute__((ext_vector_type(16))) __bf16          bf16x16;
typedef __attribute__((ext_vector_type(8)))  float           f32x8;
typedef __attribute__((ext_vector_type(8)))  unsigned short  us8;
typedef __attribute__((ext_vector_type(16))) unsigned short  us16;
typedef __attribute__((ext_vector_type(4)))  unsigned int    u32x4;
typedef __attribute__((ext_vector_type(8)))  int             i32x8;
typedef __attribute__((ext_vector_type(4)))  int             i32x4;

__device__ __forceinline__ unsigned short f2bf(float x) {
  unsigned int u = __builtin_bit_cast(unsigned int, x);
  unsigned int r = (u + 0x7FFFu + ((u >> 16) & 1u)) >> 16;  // RNE
  return (unsigned short)r;
}

// Low 32 bits of a flat shared-space pointer = LDS byte address.
__device__ __forceinline__ unsigned lds_addr32(const void* p) {
  return (unsigned)(uintptr_t)p;
}

// CDNA5 async DMA: memory -> LDS, 16B per lane, tracked by ASYNCcnt.
__device__ __forceinline__ void async_load_b128(unsigned lds_off, const void* gaddr) {
  asm volatile("global_load_async_to_lds_b128 %0, %1, off"
               :: "v"(lds_off), "v"(gaddr) : "memory");
}
__device__ __forceinline__ void wait_asynccnt0() {
  asm volatile("s_wait_asynccnt 0" ::: "memory");
}
__device__ __forceinline__ void wait_tensorcnt0() {
#if __has_builtin(__builtin_amdgcn_s_wait_tensorcnt)
  __builtin_amdgcn_s_wait_tensorcnt(0);
#else
  asm volatile("s_wait_tensorcnt 0" ::: "memory");
#endif
}

// ---------------------------------------------------------------------------
// Tensor Data Mover: one descriptor-driven DMA moves a 128x32 bf16 tile
// (row stride ldElems) into LDS at the padded 80B pitch.
// D# per CDNA5 ISA ch.8: group0 = {count, lds_addr, global_addr, type=2},
// group1 = {data_size=2B, pad_enable, pad_interval=3 (64B), pad_amount=3 (16B),
//           tensor_dim0/1 (large), tile 32x128, dim0 stride}.
// This toolchain's builtin takes 6 args (g0, g1, g2, g3, extra group, cpol).
// ---------------------------------------------------------------------------
#if __has_builtin(__builtin_amdgcn_tensor_load_to_lds)
#define HAVE_TDM 1
#endif

__device__ __forceinline__ void tdm_load_tile(const unsigned short* gsrc,
                                              int ldElems, unsigned ldsAddr)
{
#ifdef HAVE_TDM
  unsigned long long ga = (unsigned long long)(uintptr_t)gsrc;
  u32x4 g0;
  g0[0] = 1u;                                              // count=1, user mode
  g0[1] = ldsAddr;                                         // LDS byte address
  g0[2] = (unsigned)ga;                                    // global_addr[31:0]
  g0[3] = (unsigned)((ga >> 32) & 0x1FFFFFFu) | 0x80000000u; // addr[56:32]|type=2
  i32x8 g1;
  g1[0] = (int)((1u << 16) | (1u << 20) | (3u << 22) | (3u << 25));
  //        data_size=2B | pad_enable | pad every 16 DW (64B) | pad 4 DW (16B)
  g1[1] = 0;                     // abar addr=0, tensor_dim0[15:0]=0
  g1[2] = 16;                    // tensor_dim0=1<<20 (hi), tensor_dim1 lo=0
  g1[3] = 16 | (TKS << 16);      // tensor_dim1=1<<20 (hi) | tile_dim0=32
  g1[4] = 128;                   // tile_dim1=128, tile_dim2=0
  g1[5] = ldElems;               // tensor_dim0_stride[31:0]
  g1[6] = 0;                     // stride0 hi | stride1 lo
  g1[7] = 0;
  i32x4 z4 = {0, 0, 0, 0};
  i32x8 z8 = {0, 0, 0, 0, 0, 0, 0, 0};
  __builtin_amdgcn_tensor_load_to_lds(g0, g1, z4, z4, z8, 0);
#else
  // Fallback: per-lane async DMA (wave 0 covers the whole tile, 8 rows/pass).
  int lane = threadIdx.x & 31;
  for (int c = lane; c < 128 * TKS / 8; c += 32) {
    int r = c >> 2, c8 = (c & 3) * 8;
    async_load_b128(ldsAddr + (unsigned)(r * KPAD + c8) * 2u,
                    gsrc + (size_t)r * ldElems + c8);
  }
#endif
}

// Block-wide sum; valid on thread 0. Safe to call repeatedly (trailing barrier).
__device__ __forceinline__ float block_reduce_sum(float v) {
  __shared__ float red[8];
  if (threadIdx.x < 8) red[threadIdx.x] = 0.f;
  __syncthreads();
#pragma unroll
  for (int off = 16; off > 0; off >>= 1) v += __shfl_down(v, off, 32);
  if ((threadIdx.x & 31) == 0) red[threadIdx.x >> 5] = v;
  __syncthreads();
  float r = 0.f;
  if (threadIdx.x < 32) {
    r = (threadIdx.x < 8) ? red[threadIdx.x] : 0.f;
#pragma unroll
    for (int off = 4; off > 0; off >>= 1) r += __shfl_down(r, off, 32);
  }
  __syncthreads();
  return r;
}

// ---------------------------------------------------------------------------
// Shared 128x128-tile bf16 GEMM core. 256 threads = 8 waves (2x4 tiling),
// each wave owns a 32x64 sub-tile = 2x4 v_wmma_f32_16x16x32_bf16 accumulators.
// A is [row][k] row-major (lda), B is PRE-TRANSPOSED: [col][k] row-major (ldb),
// i.e. C[r][c] += sum_k A[r][k] * B[c][k].
// Fragment layouts per CDNA5 ISA 7.12.2:
//   A 16x32: lanes 0-15 M=0..15 K-halves {0..7,16..23}; lanes 16-31 {8..15,24..31}
//   B 32x16: lanes 0-15 N, K-halves {0..7,16..23}; lanes 16-31 {8..15,24..31}
// ---------------------------------------------------------------------------
__device__ __forceinline__ void issue_tile(
    const unsigned short* __restrict__ A, int lda,
    const unsigned short* __restrict__ B, int ldb,
    int rowBase, int colBase, int k0,
    unsigned short* Alds, unsigned short* Blds, int tid)
{
  // A tile: one TDM descriptor DMA, issued by wave 0 only (EXEC-independent op).
  if (tid < 32)
    tdm_load_tile(&A[(size_t)rowBase * lda + k0], lda, lds_addr32(Alds));
  // B tile: 512 16B per-lane async chunks across all 256 threads.
#pragma unroll
  for (int it = 0; it < 2; ++it) {
    int c  = tid + it * 256;
    int r  = c >> 2;
    int c8 = (c & 3) * 8;
    async_load_b128(lds_addr32(&Blds[r * KPAD + c8]),
                    &B[(size_t)(colBase + r) * ldb + k0 + c8]);
  }
}

__device__ __forceinline__ void gemm_tile_bf16(
    const unsigned short* __restrict__ A, int lda,
    const unsigned short* __restrict__ B, int ldb,
    int rowBase, int colBase, int K, f32x8 acc[2][4])
{
  __shared__ __align__(16) unsigned short Alds[2][128 * KPAD];
  __shared__ __align__(16) unsigned short Blds[2][128 * KPAD];
  const int tid  = threadIdx.x;
  const int lane = tid & 31;
  const int wid  = tid >> 5;
  const int wm   = wid >> 1;            // 0..3 -> row 32*wm
  const int wn   = wid & 1;             // 0..1 -> col 64*wn
  const int mrow = lane & 15;
  const int aK   = (lane & 16) ? 8 : 0;   // A K-group offset
  const int bK   = (lane & 16) ? 8 : 0;   // B K-group offset

  const f32x8 z = {0.f,0.f,0.f,0.f,0.f,0.f,0.f,0.f};
#pragma unroll
  for (int mi = 0; mi < 2; ++mi)
#pragma unroll
    for (int ni = 0; ni < 4; ++ni) acc[mi][ni] = z;

  const int nt = K / TKS;
  issue_tile(A, lda, B, ldb, rowBase, colBase, 0, Alds[0], Blds[0], tid);

  for (int t = 0; t < nt; ++t) {
    wait_asynccnt0();     // own tile-t B chunks retired to LDS
    wait_tensorcnt0();    // wave0's tile-t TDM A transfer complete
    __syncthreads();      // everyone's tile-t data visible; prev reads of nxt done
    if (t + 1 < nt)       // stream tile t+1 into the other buffer (overlaps WMMA)
      issue_tile(A, lda, B, ldb, rowBase, colBase, (t + 1) * TKS,
                 Alds[(t + 1) & 1], Blds[(t + 1) & 1], tid);

    const unsigned short* Ab = Alds[t & 1];
    const unsigned short* Bb = Blds[t & 1];
    bf16x16 afr[2], bfr[4];
#pragma unroll
    for (int mi = 0; mi < 2; ++mi) {
      int m = wm * 32 + mi * 16 + mrow;
      us8 lo = *(const us8*)&Ab[m * KPAD + aK];
      us8 hi = *(const us8*)&Ab[m * KPAD + 16 + aK];
      us16 tv;
#pragma unroll
      for (int q = 0; q < 8; ++q) { tv[q] = lo[q]; tv[8 + q] = hi[q]; }
      afr[mi] = __builtin_bit_cast(bf16x16, tv);
    }
#pragma unroll
    for (int ni = 0; ni < 4; ++ni) {
      int n = wn * 64 + ni * 16 + mrow;
      us8 lo = *(const us8*)&Bb[n * KPAD + bK];
      us8 hi = *(const us8*)&Bb[n * KPAD + bK + 16];
      us16 tv;
#pragma unroll
      for (int q = 0; q < 8; ++q) { tv[q] = lo[q]; tv[8 + q] = hi[q]; }
      bfr[ni] = __builtin_bit_cast(bf16x16, tv);
    }
#pragma unroll
    for (int mi = 0; mi < 2; ++mi)
#pragma unroll
      for (int ni = 0; ni < 4; ++ni)
        acc[mi][ni] = __builtin_amdgcn_wmma_f32_16x16x32_bf16(
            false, afr[mi], false, bfr[ni], (short)0, acc[mi][ni], false, false);
  }
  __syncthreads();
}

// C-fragment element -> (local row, local col) in the 128x128 tile.
#define EPI_PREAMBLE                                        \
  const int lane = threadIdx.x & 31;                        \
  const int wid  = threadIdx.x >> 5;                        \
  const int wm   = wid >> 1, wn = wid & 1;                  \
  const int rO   = (lane & 16) ? 8 : 0;                     \
  const int cO   = lane & 15;
#define EPI_LI (wm * 32 + mi * 16 + e + rO)
#define EPI_LJ (wn * 64 + ni * 16 + cO)

// ---------------------------------------------------------------------------
__global__ void zero_kernel(float* __restrict__ p, int n) {
  int i = blockIdx.x * 256 + threadIdx.x;
  if (i < n) p[i] = 0.f;
}

// Gather rows -> bf16 embeddings, fp32 row energies, and G += E^T E (ortho).
__global__ void __launch_bounds__(128)
gather_kernel(const int* __restrict__ idx, const float* __restrict__ embw,
              unsigned short* __restrict__ embsBf, float* __restrict__ energy,
              float* __restrict__ G)
{
  __shared__ float rowv[D_EMB];
  __shared__ float red[D_EMB];
  __shared__ float Gl[D_EMB * D_EMB];
  const int t = threadIdx.x;
  for (int i = t; i < D_EMB * D_EMB; i += 128) Gl[i] = 0.f;
  const int row0 = blockIdx.x * 64;
  for (int r = 0; r < 64; ++r) {
    int row = row0 + r;
    int g = idx[row];
    float w = embw[(size_t)g * D_EMB + t];
    __syncthreads();
    rowv[t] = w;
    red[t]  = w * w;
    embsBf[(size_t)row * D_EMB + t] = f2bf(w);
    __syncthreads();
#pragma unroll
    for (int s = 64; s > 0; s >>= 1) {
      if (t < s) red[t] += red[t + s];
      __syncthreads();
    }
    if (t == 0) energy[row] = sqrtf(red[0]);
    for (int a = 0; a < D_EMB; ++a) Gl[a * D_EMB + t] += rowv[a] * w;  // col t
  }
  __syncthreads();
  for (int i = t; i < D_EMB * D_EMB; i += 128) atomicAdd(&G[i], Gl[i]);
}

// trans -> bf16 copy + bf16 transposed copy (LDS tile transpose).
__global__ void __launch_bounds__(256)
trans_prep_kernel(const float* __restrict__ trans,
                  unsigned short* __restrict__ tbf,
                  unsigned short* __restrict__ tTbf)
{
  __shared__ float tile[64][65];
  const int c0 = blockIdx.x * 64, r0 = blockIdx.y * 64;
  const int tc = threadIdx.x & 63, tr = threadIdx.x >> 6;
#pragma unroll
  for (int rr = 0; rr < 16; ++rr) {
    int r = tr + rr * 4;
    float v = trans[(size_t)(r0 + r) * N_PT + c0 + tc];
    tbf[(size_t)(r0 + r) * N_PT + c0 + tc] = f2bf(v);
    tile[r][tc] = v;
  }
  __syncthreads();
#pragma unroll
  for (int rr = 0; rr < 16; ++rr) {
    int r = tr + rr * 4;  // row in T^T tile
    tTbf[(size_t)(c0 + r) * N_PT + r0 + tc] = f2bf(tile[tc][r]);
  }
}

__global__ void __launch_bounds__(256)
rowsum_kernel(const float* __restrict__ trans, float* __restrict__ rs) {
  float s = 0.f;
  for (int j = threadIdx.x; j < N_PT; j += 256)
    s += trans[(size_t)blockIdx.x * N_PT + j];
  s = block_reduce_sum(s);
  if (threadIdx.x == 0) rs[blockIdx.x] = s;
}

__global__ void __launch_bounds__(256)
colsum_kernel(const float* __restrict__ trans, float* __restrict__ cs) {
  int j = blockIdx.x * 256 + threadIdx.x;
  float s = 0.f;
  for (int i = 0; i < N_PT; ++i) s += trans[(size_t)i * N_PT + j];
  cs[j] = s;
}

// cost = 1 - exp(-5*(1 - gram/denom)); fused: bf16 store, f[r]+=mu[c]*cost^2,
// reg += (cost - ref)^2 * exp(-ref).
__global__ void __launch_bounds__(256)
cost_self_kernel(const unsigned short* __restrict__ embsBf,
                 const float* __restrict__ energy, const float* __restrict__ mu,
                 const float* __restrict__ costRef,
                 unsigned short* __restrict__ costOut,
                 float* __restrict__ fvec, float* __restrict__ regAcc)
{
  f32x8 acc[2][4];
  const int ti = blockIdx.y * 128, tj = blockIdx.x * 128;
  gemm_tile_bf16(embsBf, D_EMB, embsBf, D_EMB, ti, tj, D_EMB, acc);
  __shared__ float floc[128];
  if (threadIdx.x < 128) floc[threadIdx.x] = 0.f;
  __syncthreads();
  EPI_PREAMBLE
  float regp = 0.f;
#pragma unroll
  for (int mi = 0; mi < 2; ++mi)
#pragma unroll
    for (int ni = 0; ni < 4; ++ni)
#pragma unroll
      for (int e = 0; e < 8; ++e) {
        int li = EPI_LI, lj = EPI_LJ;
        int i = ti + li, j = tj + lj;
        float gram  = acc[mi][ni][e];
        float denom = energy[i] * energy[j] + EPSV;
        float cost  = 1.f - __expf(-5.f * (1.f - gram / denom));
        costOut[(size_t)i * N_PT + j] = f2bf(cost);
        float cr = costRef[(size_t)i * N_PT + j];
        float dd = cost - cr;
        regp += dd * dd * __expf(-cr);
        atomicAdd(&floc[li], cost * cost * mu[j]);   // ds_add_f32
      }
  float rs = block_reduce_sum(regp);
  if (threadIdx.x == 0) atomicAdd(regAcc, rs);
  __syncthreads();
  if (threadIdx.x < 128) atomicAdd(&fvec[ti + threadIdx.x], floc[threadIdx.x]);
}

// cost_st never materialized: d_w += <1 - exp(-(1 - gram/denom)), trans>.
__global__ void __launch_bounds__(256)
cost_mutual_kernel(const unsigned short* __restrict__ e1,
                   const unsigned short* __restrict__ e2,
                   const float* __restrict__ en1, const float* __restrict__ en2,
                   const float* __restrict__ trans, float* __restrict__ dwAcc)
{
  f32x8 acc[2][4];
  const int ti = blockIdx.y * 128, tj = blockIdx.x * 128;
  gemm_tile_bf16(e1, D_EMB, e2, D_EMB, ti, tj, D_EMB, acc);
  EPI_PREAMBLE
  float p = 0.f;
#pragma unroll
  for (int mi = 0; mi < 2; ++mi)
#pragma unroll
    for (int ni = 0; ni < 4; ++ni)
#pragma unroll
      for (int e = 0; e < 8; ++e) {
        int i = ti + EPI_LI, j = tj + EPI_LJ;
        float denom = en1[i] * en2[j] + EPSV;
        float cost  = 1.f - __expf(-(1.f - acc[mi][ni][e] / denom));
        p += cost * trans[(size_t)i * N_PT + j];
      }
  float s = block_reduce_sum(p);
  if (threadIdx.x == 0) atomicAdd(dwAcc, s);
}

// M2 = T @ cost_t  (cost_t symmetric: B[col][k] = cost_t row `col`).
__global__ void __launch_bounds__(256)
gemm_store_kernel(const unsigned short* __restrict__ A,
                  const unsigned short* __restrict__ Bt,
                  float* __restrict__ Cout)
{
  f32x8 acc[2][4];
  const int ti = blockIdx.y * 128, tj = blockIdx.x * 128;
  gemm_tile_bf16(A, N_PT, Bt, N_PT, ti, tj, N_PT, acc);
  EPI_PREAMBLE
#pragma unroll
  for (int mi = 0; mi < 2; ++mi)
#pragma unroll
    for (int ni = 0; ni < 4; ++ni)
#pragma unroll
      for (int e = 0; e < 8; ++e) {
        int i = ti + EPI_LI, j = tj + EPI_LJ;
        Cout[(size_t)i * N_PT + j] = acc[mi][ni][e];
      }
}

// dot += <cost_s @ T, M2>  (B = T^T pre-transposed copy).
__global__ void __launch_bounds__(256)
gemm_dot_kernel(const unsigned short* __restrict__ A,
                const unsigned short* __restrict__ Bt,
                const float* __restrict__ M2, float* __restrict__ dotAcc)
{
  f32x8 acc[2][4];
  const int ti = blockIdx.y * 128, tj = blockIdx.x * 128;
  gemm_tile_bf16(A, N_PT, Bt, N_PT, ti, tj, N_PT, acc);
  EPI_PREAMBLE
  float p = 0.f;
#pragma unroll
  for (int mi = 0; mi < 2; ++mi)
#pragma unroll
    for (int ni = 0; ni < 4; ++ni)
#pragma unroll
      for (int e = 0; e < 8; ++e) {
        int i = ti + EPI_LI, j = tj + EPI_LJ;
        p += acc[mi][ni][e] * M2[(size_t)i * N_PT + j];
      }
  float s = block_reduce_sum(p);
  if (threadIdx.x == 0) atomicAdd(dotAcc, s);
}

__global__ void __launch_bounds__(256)
final_kernel(const float* __restrict__ f1, const float* __restrict__ f2,
             const float* __restrict__ rs, const float* __restrict__ cs,
             const float* __restrict__ G1, const float* __restrict__ G2,
             const float* __restrict__ scal, float* __restrict__ out)
{
  float p1 = 0.f, p2 = 0.f, po = 0.f;
  for (int i = threadIdx.x; i < N_PT; i += 256) {
    p1 += f1[i] * rs[i];
    p2 += f2[i] * cs[i];
  }
  for (int i = threadIdx.x; i < D_EMB * D_EMB; i += 256) {
    float d = (i % (D_EMB + 1)) == 0 ? 1.f : 0.f;
    float a = G1[i] - d, b = G2[i] - d;
    po += a * a + b * b;
  }
  float s1 = block_reduce_sum(p1);
  float s2 = block_reduce_sum(p2);
  float so = block_reduce_sum(po);
  if (threadIdx.x == 0) {
    out[0] = s1 + s2 - 2.f * scal[3];   // d_gw
    out[1] = scal[0];                   // d_w
    out[2] = scal[1] + scal[2] + so;    // regularizer
  }
}

// ---------------------------------------------------------------------------
extern "C" void kernel_launch(void* const* d_in, const int* in_sizes, int n_in,
                              void* d_out, int out_size, void* d_ws, size_t ws_size,
                              hipStream_t stream)
{
  (void)in_sizes; (void)n_in; (void)out_size; (void)ws_size;
  const int*   index1 = (const int*)d_in[0];
  const int*   index2 = (const int*)d_in[1];
  const float* trans  = (const float*)d_in[2];
  const float* mu_s   = (const float*)d_in[3];
  const float* mu_t   = (const float*)d_in[4];
  const float* cost1  = (const float*)d_in[5];
  const float* cost2  = (const float*)d_in[6];
  const float* emb1w  = (const float*)d_in[7];
  const float* emb2w  = (const float*)d_in[8];
  float* out = (float*)d_out;

  char* ws = (char*)d_ws;
  unsigned short* embs1   = (unsigned short*)(ws);                       // 1 MB
  unsigned short* embs2   = (unsigned short*)(ws + (1ull << 20));        // 1 MB
  float*          energy1 = (float*)(ws + (2ull << 20));                 // 16 KB
  float*          energy2 = (float*)(ws + (2ull << 20) + (16u << 10));   // 16 KB
  float*          f1      = (float*)(ws + (2ull << 20) + (32u << 10));   // zero zone start
  float*          f2      = f1 + N_PT;
  float*          G1      = f2 + N_PT;
  float*          G2      = G1 + D_EMB * D_EMB;
  float*          scal    = G2 + D_EMB * D_EMB;  // [0]=d_w [1]=reg1 [2]=reg2 [3]=gw_dot
  const int nzero = N_PT + N_PT + 2 * D_EMB * D_EMB + 16;
  float*          rowsumv = scal + 16;
  float*          colsumv = rowsumv + N_PT;
  unsigned short* cost_s  = (unsigned short*)(ws + (4ull   << 20));  // 32 MB bf16
  unsigned short* cost_t  = (unsigned short*)(ws + (36ull  << 20));  // 32 MB bf16
  unsigned short* tbf     = (unsigned short*)(ws + (68ull  << 20));  // 32 MB bf16
  unsigned short* tTbf    = (unsigned short*)(ws + (100ull << 20));  // 32 MB bf16
  float*          M2      = (float*)(ws + (132ull << 20));           // 64 MB f32

  zero_kernel<<<(nzero + 255) / 256, 256, 0, stream>>>(f1, nzero);
  gather_kernel<<<N_PT / 64, 128, 0, stream>>>(index1, emb1w, embs1, energy1, G1);
  gather_kernel<<<N_PT / 64, 128, 0, stream>>>(index2, emb2w, embs2, energy2, G2);
  trans_prep_kernel<<<dim3(64, 64), 256, 0, stream>>>(trans, tbf, tTbf);
  rowsum_kernel<<<N_PT, 256, 0, stream>>>(trans, rowsumv);
  colsum_kernel<<<N_PT / 256, 256, 0, stream>>>(trans, colsumv);

  dim3 g32(32, 32);
  cost_self_kernel<<<g32, 256, 0, stream>>>(embs1, energy1, mu_s, cost1,
                                            cost_s, f1, &scal[1]);
  cost_self_kernel<<<g32, 256, 0, stream>>>(embs2, energy2, mu_t, cost2,
                                            cost_t, f2, &scal[2]);
  cost_mutual_kernel<<<g32, 256, 0, stream>>>(embs1, embs2, energy1, energy2,
                                              trans, &scal[0]);
  gemm_store_kernel<<<g32, 256, 0, stream>>>(tbf, cost_t, M2);
  gemm_dot_kernel<<<g32, 256, 0, stream>>>(cost_s, tTbf, M2, &scal[3]);
  final_kernel<<<1, 256, 0, stream>>>(f1, f2, rowsumv, colsumv, G1, G2, scal, out);
}
